// PhiMoEAttention_7516192768998
// MI455X (gfx1250) — compile-verified, matrix operand order checked by reading, not run on hardware
//
#include <hip/hip_runtime.h>

// ---------------------------------------------------------------------------
// Types for CDNA5 WMMA (wave32, v_wmma_f32_16x16x32_bf16)
// ---------------------------------------------------------------------------
typedef __attribute__((ext_vector_type(8)))  __bf16 v8bf;
typedef __attribute__((ext_vector_type(16))) __bf16 v16bf;
typedef __attribute__((ext_vector_type(8)))  float  v8f;

#define WMMA_BF16(a, b, c) \
  __builtin_amdgcn_wmma_f32_16x16x32_bf16(false, (a), false, (b), (short)0, (c), false, false)

#define AS3 __attribute__((address_space(3)))

// Async global->LDS (CDNA5 ASYNCcnt path) via inline asm; the clang builtin's
// params use HIP language address spaces that can't be spelled in source.
#if defined(__HIP_DEVICE_COMPILE__) && defined(__gfx1250__)
#define HAS_ASYNC_LDS 1
#else
#define HAS_ASYNC_LDS 0
#endif

static __device__ __forceinline__ void cp16_g2l(const unsigned short* g,
                                                unsigned short* l) {
#if HAS_ASYNC_LDS
  unsigned laddr = (unsigned)(size_t)(AS3 unsigned short*)l;  // 32-bit LDS byte addr
  asm volatile("global_load_async_to_lds_b128 %0, %1, off"
               :
               : "v"(laddr), "v"(g)
               : "memory");
#else
  *(uint4*)l = *(const uint4*)g;
#endif
}

static __device__ __forceinline__ void wait_async_all() {
#if HAS_ASYNC_LDS
  asm volatile("s_wait_asynccnt 0x0" ::: "memory");
#endif
}

static __device__ __forceinline__ unsigned short f2bfu(float f) {
  unsigned u = __builtin_bit_cast(unsigned, f);
  u = (u + 0x7fffu + ((u >> 16) & 1u)) >> 16;   // round-to-nearest-even
  return (unsigned short)u;
}

// Build a v16bf fragment from two contiguous 16-byte chunks (LDS or global).
static __device__ __forceinline__ v16bf frag_ld(const unsigned short* p0,
                                                const unsigned short* p1) {
  v8bf lo = *(const v8bf*)p0;
  v8bf hi = *(const v8bf*)p1;
  return __builtin_shufflevector(lo, hi, 0,1,2,3,4,5,6,7,8,9,10,11,12,13,14,15);
}

// ---------------------------------------------------------------------------
// Problem constants
// ---------------------------------------------------------------------------
#define BATCH 2
#define SEQ   2048
#define HID   4096
#define NQH   32
#define NKVH  8
#define HD    128
#define QKV_N 6144
#define LOG_THETA 13.815510557964274f   // ln(1e6)

// ---------------------------------------------------------------------------
// fp32 -> bf16 conversion (vectorized x4)
// ---------------------------------------------------------------------------
__global__ void k_f32_to_bf16(const float* __restrict__ in,
                              unsigned short* __restrict__ out, int n4) {
  int i = blockIdx.x * blockDim.x + threadIdx.x;
  if (i >= n4) return;
  float4 v = ((const float4*)in)[i];
  ushort4 o;
  o.x = f2bfu(v.x); o.y = f2bfu(v.y); o.z = f2bfu(v.z); o.w = f2bfu(v.w);
  ((ushort4*)out)[i] = o;
}

// ---------------------------------------------------------------------------
// Generic bf16 GEMM:  C[M,N] (fp32) = A[M,K] * B[N,K]^T   (K contiguous both)
// Block: 256 thr (8 waves), 128x128 C tile, K step 32, double-buffered LDS
// fed by async global->LDS DMA. Each wave: 64x32 = 4x2 WMMA accumulators.
// ---------------------------------------------------------------------------
__global__ __launch_bounds__(256) void k_gemm_bf16(
    const unsigned short* __restrict__ A,
    const unsigned short* __restrict__ B,
    float* __restrict__ C, int M, int N, int K) {
  __shared__ __align__(16) unsigned short lA[2][128 * 32];
  __shared__ __align__(16) unsigned short lB[2][128 * 32];

  const int tid  = threadIdx.x;
  const int lane = tid & 31;
  const int w    = tid >> 5;      // wave 0..7
  const int wr   = w & 1;         // M half (64 rows each)
  const int wc   = w >> 1;        // N quarter (32 cols each)
  const int mBase = blockIdx.y * 128;
  const int nBase = blockIdx.x * 128;
  const int lm = lane & 15;
  const int g8 = (lane >> 4) * 8;

  // copy slots: this thread owns chunks tid and tid+256 of each 512-chunk tile
  const int r0 = tid >> 2,          o0 = (tid & 3) * 8;
  const int r1 = (tid + 256) >> 2,  o1 = (tid & 3) * 8;
  const unsigned short* gA0 = A + (size_t)(mBase + r0) * K + o0;
  const unsigned short* gA1 = A + (size_t)(mBase + r1) * K + o1;
  const unsigned short* gB0 = B + (size_t)(nBase + r0) * K + o0;
  const unsigned short* gB1 = B + (size_t)(nBase + r1) * K + o1;
  const int lo0 = r0 * 32 + o0, lo1 = r1 * 32 + o1;

  v8f acc[4][2];
#pragma unroll
  for (int a = 0; a < 4; ++a)
#pragma unroll
    for (int bb = 0; bb < 2; ++bb)
#pragma unroll
      for (int e = 0; e < 8; ++e) acc[a][bb][e] = 0.0f;

  // prologue: DMA tile 0 into buffer 0
  cp16_g2l(gA0, &lA[0][lo0]);  cp16_g2l(gA1, &lA[0][lo1]);
  cp16_g2l(gB0, &lB[0][lo0]);  cp16_g2l(gB1, &lB[0][lo1]);
  gA0 += 32; gA1 += 32; gB0 += 32; gB1 += 32;

  const int nk = K >> 5;
  for (int i = 0; i < nk; ++i) {
    const int p = i & 1;
    wait_async_all();        // this thread's tile-i DMA done
    __syncthreads();         // everyone's tile-i data visible; prev reads done
    if (i + 1 < nk) {        // DMA tile i+1 into the other buffer
      cp16_g2l(gA0, &lA[p ^ 1][lo0]);  cp16_g2l(gA1, &lA[p ^ 1][lo1]);
      cp16_g2l(gB0, &lB[p ^ 1][lo0]);  cp16_g2l(gB1, &lB[p ^ 1][lo1]);
      gA0 += 32; gA1 += 32; gB0 += 32; gB1 += 32;
    }

    v16bf afr[4], bfr[2];
#pragma unroll
    for (int mt = 0; mt < 4; ++mt) {
      const unsigned short* r = &lA[p][(wr * 64 + mt * 16 + lm) * 32];
      afr[mt] = frag_ld(r + g8, r + 16 + g8);
    }
#pragma unroll
    for (int nt = 0; nt < 2; ++nt) {
      const unsigned short* r = &lB[p][(wc * 32 + nt * 16 + lm) * 32];
      bfr[nt] = frag_ld(r + g8, r + 16 + g8);
    }
#pragma unroll
    for (int mt = 0; mt < 4; ++mt)
#pragma unroll
      for (int nt = 0; nt < 2; ++nt)
        acc[mt][nt] = WMMA_BF16(afr[mt], bfr[nt], acc[mt][nt]);
  }

  const int lg = lane >> 4;
#pragma unroll
  for (int mt = 0; mt < 4; ++mt)
#pragma unroll
    for (int nt = 0; nt < 2; ++nt)
#pragma unroll
      for (int r = 0; r < 8; ++r) {
        int row = mBase + wr * 64 + mt * 16 + r + lg * 8;
        int col = nBase + wc * 32 + nt * 16 + lm;
        C[(size_t)row * N + col] = acc[mt][nt][r];
      }
}

// ---------------------------------------------------------------------------
// RoPE for Q and K: reads fp32 qkv, writes head-major bf16
//   Qr: [B, 32, S, 128], Kr: [B, 8, S, 128]
// ---------------------------------------------------------------------------
__global__ void k_rope_qk(const float* __restrict__ qkv,
                          const int* __restrict__ pos,
                          unsigned short* __restrict__ Qr,
                          unsigned short* __restrict__ Kr) {
  int idx = blockIdx.x * blockDim.x + threadIdx.x;
  const int total = BATCH * SEQ * (NQH + NKVH) * 64;
  if (idx >= total) return;
  int bs  = idx / ((NQH + NKVH) * 64);
  int rem = idx - bs * ((NQH + NKVH) * 64);
  int h = rem >> 6;
  int i = rem & 63;
  int b = bs / SEQ, s = bs - b * SEQ;
  float p = (float)pos[bs];
  float freq = p * __expf(-(float)i * (LOG_THETA / 64.0f));
  float c = __cosf(freq), sn = __sinf(freq);
  if (h < NQH) {
    size_t base = (size_t)bs * QKV_N + h * HD;
    float x1 = qkv[base + i], x2 = qkv[base + 64 + i];
    size_t ob = ((size_t)(b * NQH + h) * SEQ + s) * HD;
    Qr[ob + i]      = f2bfu(x1 * c - x2 * sn);
    Qr[ob + 64 + i] = f2bfu(x2 * c + x1 * sn);
  } else {
    int g = h - NQH;
    size_t base = (size_t)bs * QKV_N + NQH * HD + g * HD;
    float x1 = qkv[base + i], x2 = qkv[base + 64 + i];
    size_t ob = ((size_t)(b * NKVH + g) * SEQ + s) * HD;
    Kr[ob + i]      = f2bfu(x1 * c - x2 * sn);
    Kr[ob + 64 + i] = f2bfu(x2 * c + x1 * sn);
  }
}

// V repack: fp32 qkv -> bf16 Vc[B, 8, S, 128]
__global__ void k_pack_v(const float* __restrict__ qkv,
                         unsigned short* __restrict__ Vc) {
  int idx = blockIdx.x * blockDim.x + threadIdx.x;
  const int total = BATCH * SEQ * NKVH * HD;
  if (idx >= total) return;
  int bs = idx >> 10;
  int r  = idx & 1023;
  int g = r >> 7, d = r & 127;
  int b = bs / SEQ, s = bs - b * SEQ;
  Vc[((size_t)(b * NKVH + g) * SEQ + s) * HD + d] =
      f2bfu(qkv[(size_t)bs * QKV_N + (NQH + NKVH) * HD + g * HD + d]);
}

// ---------------------------------------------------------------------------
// Causal GQA flash attention, bf16 WMMA, fp32 online softmax.
// grid = (S/64, 32 q-heads, B); block = 128 (4 waves); wave owns 16 q rows.
// Key tile = 32 keys. LDS: K[32][128] (async DMA), Vt[128][32], P[4][16][32].
// ---------------------------------------------------------------------------
__global__ __launch_bounds__(128) void k_attn(
    const unsigned short* __restrict__ Qr,
    const unsigned short* __restrict__ Kr,
    const unsigned short* __restrict__ Vc,
    unsigned short* __restrict__ Out) {
  __shared__ __align__(16) unsigned short lK[32 * 128];
  __shared__ __align__(16) unsigned short lVt[128 * 32];
  __shared__ __align__(16) unsigned short lP[4 * 16 * 32];

  const int b = blockIdx.z, h = blockIdx.y, qt = blockIdx.x;
  const int g = h >> 2;                      // kv head (R = 4)
  const int tid = threadIdx.x;
  const int lane = tid & 31;
  const int w = tid >> 5;
  const int q0 = qt * 64 + w * 16;
  const int lm = lane & 15;
  const int lg = lane >> 4;
  const int g8 = lg * 8;
  const float scale = 0.08838834764831845f;  // 1/sqrt(128)

  // Q fragments (registers for whole kernel)
  v16bf qf[4];
  {
    const size_t qbase = ((size_t)(b * NQH + h) * SEQ + q0 + lm) * HD;
#pragma unroll
    for (int dc = 0; dc < 4; ++dc) {
      const unsigned short* r = &Qr[qbase + dc * 32];
      qf[dc] = frag_ld(r + g8, r + 16 + g8);
    }
  }

  v8f o[8];
#pragma unroll
  for (int dt = 0; dt < 8; ++dt)
#pragma unroll
    for (int e = 0; e < 8; ++e) o[dt][e] = 0.0f;
  float mS[8], lS[8];
#pragma unroll
  for (int r = 0; r < 8; ++r) { mS[r] = -3.0e38f; lS[r] = 0.0f; }

  const size_t kvbase = (size_t)(b * NKVH + g) * SEQ;
  const int nkt = 2 * qt + 2;  // causal: last needed key tile

  for (int kt = 0; kt < nkt; ++kt) {
    const int kbase = kt * 32;
    __syncthreads();           // previous iteration's LDS reads complete
    // K tile [key][d]: async DMA, 4 chunks per thread (constant trip count)
#pragma unroll
    for (int i = 0; i < 4; ++i) {
      int c = tid + i * 128;
      int key = c >> 4, d8 = (c & 15) * 8;
      cp16_g2l(&Kr[(kvbase + kbase + key) * HD + d8], &lK[key * 128 + d8]);
    }
    // V tile transposed into [d][key] (needs VGPR bounce for transpose)
#pragma unroll
    for (int i = 0; i < 4; ++i) {
      int c = tid + i * 128;
      int key = c >> 4, d8 = (c & 15) * 8;
      union { uint4 u; unsigned short s[8]; } uu;
      uu.u = *(const uint4*)&Vc[(kvbase + kbase + key) * HD + d8];
#pragma unroll
      for (int j = 0; j < 8; ++j) lVt[(d8 + j) * 32 + key] = uu.s[j];
    }
    wait_async_all();
    __syncthreads();

    // S = Q*K^T for two 16-key sub-tiles
    float sv[2][8];
#pragma unroll
    for (int n = 0; n < 2; ++n) {
      v8f sa;
#pragma unroll
      for (int e = 0; e < 8; ++e) sa[e] = 0.0f;
#pragma unroll
      for (int dc = 0; dc < 4; ++dc) {
        const unsigned short* r = &lK[(n * 16 + lm) * 128 + dc * 32];
        v16bf kf = frag_ld(r + g8, r + 16 + g8);
        sa = WMMA_BF16(qf[dc], kf, sa);
      }
      int col = kbase + n * 16 + lm;
#pragma unroll
      for (int r = 0; r < 8; ++r) {
        int row = q0 + r + lg * 8;
        float v = sa[r] * scale;
        sv[n][r] = (col <= row) ? v : -3.0e38f;
      }
    }

    // online softmax per query row (16-lane N-group reductions)
#pragma unroll
    for (int r = 0; r < 8; ++r) {
      float mt = fmaxf(sv[0][r], sv[1][r]);
#pragma unroll
      for (int off = 8; off > 0; off >>= 1)
        mt = fmaxf(mt, __shfl_xor(mt, off, 32));
      float mn = fmaxf(mS[r], mt);
      float al = __expf(mS[r] - mn);
      mS[r] = mn;
      float p0 = __expf(sv[0][r] - mn);
      float p1 = __expf(sv[1][r] - mn);
      int prow = (r + lg * 8) * 32;
      lP[w * 512 + prow + lm]      = f2bfu(p0);
      lP[w * 512 + prow + 16 + lm] = f2bfu(p1);
      float ps = p0 + p1;
#pragma unroll
      for (int off = 8; off > 0; off >>= 1)
        ps += __shfl_xor(ps, off, 32);
      lS[r] = lS[r] * al + ps;
#pragma unroll
      for (int dt = 0; dt < 8; ++dt) o[dt][r] *= al;
    }

    // O += P * V
    {
      const unsigned short* pr = &lP[w * 512 + lm * 32];
      v16bf pf = frag_ld(pr + g8, pr + 16 + g8);
#pragma unroll
      for (int dt = 0; dt < 8; ++dt) {
        const unsigned short* r = &lVt[(dt * 16 + lm) * 32];
        v16bf vf = frag_ld(r + g8, r + 16 + g8);
        o[dt] = WMMA_BF16(pf, vf, o[dt]);
      }
    }
  }

  // epilogue: divide by l, store bf16 into [b, s, h*128 + d]
#pragma unroll
  for (int r = 0; r < 8; ++r) {
    float inv = 1.0f / lS[r];
    size_t rowb = ((size_t)b * SEQ + q0 + r + lg * 8) * HID + h * HD;
#pragma unroll
    for (int dt = 0; dt < 8; ++dt)
      Out[rowb + dt * 16 + lm] = f2bfu(o[dt][r] * inv);
  }
}

// ---------------------------------------------------------------------------
// launch
// ---------------------------------------------------------------------------
extern "C" void kernel_launch(void* const* d_in, const int* in_sizes, int n_in,
                              void* d_out, int out_size, void* d_ws, size_t ws_size,
                              hipStream_t stream) {
  const float* hidden = (const float*)d_in[0];
  const int*   pos    = (const int*)d_in[1];
  const float* w_qkv  = (const float*)d_in[2];
  const float* w_o    = (const float*)d_in[3];
  float* out = (float*)d_out;

  const size_t M = (size_t)BATCH * SEQ;          // 4096 token rows
  char* ws = (char*)d_ws;
  size_t off = 0;
  unsigned short* hid_bf  = (unsigned short*)(ws + off); off += M * HID * 2;
  unsigned short* wqkv_bf = (unsigned short*)(ws + off); off += (size_t)QKV_N * HID * 2;
  unsigned short* wo_bf   = (unsigned short*)(ws + off); off += (size_t)HID * HID * 2;
  float*          qkv_f   = (float*)(ws + off);          off += M * QKV_N * 4;
  unsigned short* Qr      = (unsigned short*)(ws + off); off += (size_t)BATCH * NQH * SEQ * HD * 2;
  unsigned short* Kr      = (unsigned short*)(ws + off); off += (size_t)BATCH * NKVH * SEQ * HD * 2;
  unsigned short* Vc      = (unsigned short*)(ws + off); off += (size_t)BATCH * NKVH * SEQ * HD * 2;
  unsigned short* attn_bf = (unsigned short*)(ws + off); off += M * HID * 2;
  (void)ws_size; (void)in_sizes; (void)n_in; (void)out_size;

  // 1) fp32 -> bf16 operand conversion
  {
    int n4 = (int)(M * HID / 4);
    k_f32_to_bf16<<<(n4 + 255) / 256, 256, 0, stream>>>(hidden, hid_bf, n4);
    n4 = (int)((size_t)QKV_N * HID / 4);
    k_f32_to_bf16<<<(n4 + 255) / 256, 256, 0, stream>>>(w_qkv, wqkv_bf, n4);
    n4 = (int)((size_t)HID * HID / 4);
    k_f32_to_bf16<<<(n4 + 255) / 256, 256, 0, stream>>>(w_o, wo_bf, n4);
  }

  // 2) QKV projection: qkv_f[M, 6144] = hidden * w_qkv^T
  k_gemm_bf16<<<dim3(QKV_N / 128, (unsigned)(M / 128)), 256, 0, stream>>>(
      hid_bf, wqkv_bf, qkv_f, (int)M, QKV_N, HID);

  // 3) RoPE on Q,K (head-major repack) + V repack
  {
    int total = BATCH * SEQ * (NQH + NKVH) * 64;
    k_rope_qk<<<(total + 255) / 256, 256, 0, stream>>>(qkv_f, pos, Qr, Kr);
    total = BATCH * SEQ * NKVH * HD;
    k_pack_v<<<(total + 255) / 256, 256, 0, stream>>>(qkv_f, Vc);
  }

  // 4) causal GQA flash attention
  k_attn<<<dim3(SEQ / 64, NQH, BATCH), 128, 0, stream>>>(Qr, Kr, Vc, attn_bf);

  // 5) output projection: out[M, 4096] = attn * w_o^T (fp32 result)
  k_gemm_bf16<<<dim3(HID / 128, (unsigned)(M / 128)), 256, 0, stream>>>(
      attn_bf, wo_bf, out, (int)M, HID, HID);
}
